// SwinEncoder_25074019074398
// MI455X (gfx1250) — matrix operand order repaired and checked by reading, not compiled, not run on hardware
//
#include <hip/hip_runtime.h>
#include <hip/hip_bf16.h>
#include <math.h>

// ---------------------------------------------------------------------------
// Types for CDNA5 WMMA (wave32, 16x16x32 bf16 -> f32 accumulate)
// ---------------------------------------------------------------------------
typedef __attribute__((ext_vector_type(16))) __bf16 v16bf;
typedef __attribute__((ext_vector_type(8)))  __bf16 v8bf;
typedef __attribute__((ext_vector_type(8)))  float  v8f;

#define GT_M 128
#define GT_N 128
#define GT_K 32
#define LDSP 40   // padded LDS row stride (bf16 elems); 80B = multiple of 16B

#define EPI_NONE 0
#define EPI_BIAS 1
#define EPI_GELU 2

__device__ __forceinline__ float gelu_exact(float x) {
    return 0.5f * x * (1.0f + erff(x * 0.70710678118654752f));
}

// ---------------------------------------------------------------------------
// Weight convert + transpose: Bt[n*K + k] = (bf16) B[k*N + n]
// 32x32 LDS tile, 256 threads (32x8).
// ---------------------------------------------------------------------------
__global__ __launch_bounds__(256)
void convT_kernel(const float* __restrict__ B, __bf16* __restrict__ Bt, int K, int N)
{
    __shared__ __bf16 t[32][33];
    const int kb = blockIdx.y * 32, nb = blockIdx.x * 32;
    const int tx = threadIdx.x & 31, ty = threadIdx.x >> 5;
#pragma unroll
    for (int i = 0; i < 32; i += 8) {
        const int k = kb + ty + i, n = nb + tx;
        t[ty + i][tx] = (k < K && n < N) ? (__bf16)B[(size_t)k * N + n] : (__bf16)0.0f;
    }
    __syncthreads();
#pragma unroll
    for (int i = 0; i < 32; i += 8) {
        const int n = nb + ty + i, k = kb + tx;
        if (n < N && k < K) Bt[(size_t)n * K + k] = t[tx][ty + i];
    }
}

// ---------------------------------------------------------------------------
// Generic GEMM: C[M,N] = A[M,K] * Bt[N,K]^T (+bias, +gelu), bf16 in, f32 acc.
// A: [M][K] bf16 row-major.  Bt: [N][K] bf16 (pre-transposed weights).
// K % 32 == 0, N % 128 == 0 for this model. M edge handled.
// Block: 256 threads = 8 waves, tile 128x128, each wave 32(M)x64(N).
// Software-pipelined: next tile fetched to regs while WMMAs run.
// ---------------------------------------------------------------------------
__global__ __launch_bounds__(256)
void gemm_wmma_bf16(const __bf16* __restrict__ A, const __bf16* __restrict__ Bt,
                    const float* __restrict__ bias, void* __restrict__ Cv,
                    int M, int N, int K, int epi, int out_bf16)
{
    __shared__ __bf16 lds_a[GT_M * LDSP];   // [m][k]
    __shared__ __bf16 lds_b[GT_N * LDSP];   // [n][k]

    const int tid  = threadIdx.x;
    const int wave = tid >> 5;
    const int lane = tid & 31;
    const int l15  = lane & 15;
    const int lhi  = lane >> 4;        // 0 or 1

    const int m_blk = blockIdx.y * GT_M;
    const int n_blk = blockIdx.x * GT_N;

    // wave grid: 4 (M) x 2 (N); wave computes 32(M) x 64(N)
    const int wm = (wave >> 1) * 32;
    const int wn = (wave & 1) * 64;

    v8f acc[2][4];
#pragma unroll
    for (int mi = 0; mi < 2; ++mi)
#pragma unroll
        for (int ni = 0; ni < 4; ++ni)
#pragma unroll
            for (int r = 0; r < 8; ++r) acc[mi][ni][r] = 0.0f;

    // staging map: thread -> (row, 16-elem K half). 256 threads cover 128 rows x 32 k.
    const int row_s = tid >> 1;
    const int kh    = (tid & 1) * 16;
    const int ga    = m_blk + row_s;          // A global row (may exceed M)
    const int gb    = n_blk + row_s;          // Bt global row (always < N)
    const bool avalid = ga < M;
    const __bf16* aptr = A  + (size_t)(avalid ? ga : 0) * K + kh;
    const __bf16* bptr = Bt + (size_t)gb * K + kh;
    __bf16* as = &lds_a[row_s * LDSP + kh];
    __bf16* bs = &lds_b[row_s * LDSP + kh];

    v8bf a0, a1, b0, b1;
    const v8bf zero8 = (v8bf)(__bf16)0.0f;

    // preload tile k0 = 0
    if (avalid) { a0 = *(const v8bf*)(aptr);  a1 = *(const v8bf*)(aptr + 8); }
    else        { a0 = zero8; a1 = zero8; }
    b0 = *(const v8bf*)(bptr);  b1 = *(const v8bf*)(bptr + 8);

    for (int k0 = 0; k0 < K; k0 += GT_K) {
        // commit staged tile to LDS
        *(v8bf*)(as)     = a0;  *(v8bf*)(as + 8) = a1;
        *(v8bf*)(bs)     = b0;  *(v8bf*)(bs + 8) = b1;
        __syncthreads();

        // prefetch next tile into registers (overlaps the WMMAs below)
        if (k0 + GT_K < K) {
            const __bf16* an = aptr + k0 + GT_K;
            const __bf16* bn = bptr + k0 + GT_K;
            if (avalid) { a0 = *(const v8bf*)(an); a1 = *(const v8bf*)(an + 8); }
            b0 = *(const v8bf*)(bn);  b1 = *(const v8bf*)(bn + 8);
        }

        // per-lane fragment gathers (ISA 16-bit A 16x32 / B 32x16 layouts)
        v16bf afrag[2], bfrag[4];
#pragma unroll
        for (int mi = 0; mi < 2; ++mi) {
            const __bf16* ar = &lds_a[(wm + mi * 16 + l15) * LDSP + lhi * 8];
            v8bf lo = *(const v8bf*)(ar);        // K 0..7   (or 8..15)
            v8bf hi = *(const v8bf*)(ar + 16);   // K 16..23 (or 24..31)
#pragma unroll
            for (int i = 0; i < 8; ++i) { afrag[mi][i] = lo[i]; afrag[mi][i + 8] = hi[i]; }
        }
#pragma unroll
        for (int ni = 0; ni < 4; ++ni) {
            const __bf16* br = &lds_b[(wn + ni * 16 + l15) * LDSP + lhi * 8];
            v8bf lo = *(const v8bf*)(br);        // K 0..7   (or 8..15)
            v8bf hi = *(const v8bf*)(br + 16);   // K 16..23 (or 24..31)
#pragma unroll
            for (int i = 0; i < 8; ++i) { bfrag[ni][i] = lo[i]; bfrag[ni][i + 8] = hi[i]; }
        }

#pragma unroll
        for (int mi = 0; mi < 2; ++mi)
#pragma unroll
            for (int ni = 0; ni < 4; ++ni)
                acc[mi][ni] = __builtin_amdgcn_wmma_f32_16x16x32_bf16(
                    false, afrag[mi], false, bfrag[ni],
                    (short)0, acc[mi][ni], false, false);

        __syncthreads();
    }

    // epilogue: lane L, vgpr r holds (M = r + 8*lhi, N = l15)
    const bool full = (m_blk + GT_M) <= M;
    float* Cf = (float*)Cv;
    __bf16* Ch = (__bf16*)Cv;
#pragma unroll
    for (int mi = 0; mi < 2; ++mi) {
#pragma unroll
        for (int ni = 0; ni < 4; ++ni) {
            const int col = n_blk + wn + ni * 16 + l15;
            const float bv = (epi != EPI_NONE) ? bias[col] : 0.0f;
#pragma unroll
            for (int r = 0; r < 8; ++r) {
                const int row = m_blk + wm + mi * 16 + lhi * 8 + r;
                if (full || row < M) {
                    float v = acc[mi][ni][r] + bv;
                    if (epi == EPI_GELU) v = gelu_exact(v);
                    const size_t o = (size_t)row * N + col;
                    if (out_bf16) Ch[o] = (__bf16)v;
                    else          Cf[o] = v;
                }
            }
        }
    }
}

// ---------------------------------------------------------------------------
// LayerNorm over last dim C; one block (256 thr) per row.
// Writes fp32 and/or bf16 output (either pointer may be null).
// ---------------------------------------------------------------------------
__global__ __launch_bounds__(256)
void ln_kernel(const float* __restrict__ in, const float* __restrict__ g,
               const float* __restrict__ b, float* __restrict__ outf,
               __bf16* __restrict__ outh, int C)
{
    __shared__ float red[256];
    const int row = blockIdx.x;
    const int tid = threadIdx.x;
    const float* x = in + (size_t)row * C;

    float s = 0.0f;
    for (int c = tid; c < C; c += 256) s += x[c];
    red[tid] = s; __syncthreads();
    for (int o = 128; o > 0; o >>= 1) { if (tid < o) red[tid] += red[tid + o]; __syncthreads(); }
    const float mu = red[0] / (float)C;
    __syncthreads();

    float v = 0.0f;
    for (int c = tid; c < C; c += 256) { float d = x[c] - mu; v += d * d; }
    red[tid] = v; __syncthreads();
    for (int o = 128; o > 0; o >>= 1) { if (tid < o) red[tid] += red[tid + o]; __syncthreads(); }
    const float rstd = rsqrtf(red[0] / (float)C + 1e-5f);

    for (int c = tid; c < C; c += 256) {
        const float y = (x[c] - mu) * rstd * g[c] + b[c];
        if (outf) outf[(size_t)row * C + c] = y;
        if (outh) outh[(size_t)row * C + c] = (__bf16)y;
    }
}

// ---------------------------------------------------------------------------
// Window partition with optional shift-roll: x[H*W,C] fp32 -> win[nW*100,C] bf16
// ---------------------------------------------------------------------------
__global__ __launch_bounds__(256)
void win_part_kernel(const float* __restrict__ x, __bf16* __restrict__ win,
                     int H, int W, int C, int shift)
{
    const int total = H * W * C;
    const int i = blockIdx.x * 256 + threadIdx.x;
    if (i >= total) return;
    const int c = i % C;
    const int t = i / C;
    const int wnum = t / 100;
    const int p = t % 100;
    const int r = p / 10, cc = p % 10;
    const int Wg = W / 10;
    const int wy = wnum / Wg, wx = wnum % Wg;
    int h = wy * 10 + r, w = wx * 10 + cc;
    if (shift) { h = (h + shift) % H; w = (w + shift) % W; }
    win[i] = (__bf16)x[((size_t)h * W + w) * C + c];
}

// ---------------------------------------------------------------------------
// Window reverse (+ un-shift roll) fused with residual add:
//   x[t,c] = shortcut[t,c] + win[rev(t),c]
// ---------------------------------------------------------------------------
__global__ __launch_bounds__(256)
void win_rev_add_kernel(const float* __restrict__ win, float* __restrict__ x,
                        int H, int W, int C, int shift)
{
    const int total = H * W * C;
    const int i = blockIdx.x * 256 + threadIdx.x;
    if (i >= total) return;
    const int c = i % C;
    const int t = i / C;
    int h = t / W, w = t % W;
    if (shift) { h = (h - shift + H) % H; w = (w - shift + W) % W; }
    const int Wg = W / 10;
    const int wy = h / 10, r = h % 10;
    const int wx = w / 10, cc = w % 10;
    const int row = (wy * Wg + wx) * 100 + r * 10 + cc;
    x[i] = x[i] + win[(size_t)row * C + c];
}

// ---------------------------------------------------------------------------
// Window attention: one block per (window, head). hd == 32 for all stages.
// qkv row layout: [3C] = (s, h, d) -> s*nH*32 + h*32 + d.  Output bf16.
// ---------------------------------------------------------------------------
__device__ __forceinline__ int region_id(int h, int Hfull) {
    if (h < Hfull - 10) return 0;
    if (h < Hfull - 5)  return 1;
    return 2;
}

__global__ __launch_bounds__(128)
void attn_kernel(const float* __restrict__ qkv, const float* __restrict__ rpb,
                 __bf16* __restrict__ out, int nH, int H, int W, int shift)
{
    __shared__ float ks[100][32];
    __shared__ float vs[100][32];
    const int win  = blockIdx.x;
    const int head = blockIdx.y;
    const int C = nH * 32;
    const int tid = threadIdx.x;
    const float* base = qkv + (size_t)win * 100 * 3 * C;

    for (int i = tid; i < 100 * 32; i += 128) {
        const int r = i >> 5, d = i & 31;
        ks[r][d] = base[(size_t)r * 3 * C + (nH + head) * 32 + d];
        vs[r][d] = base[(size_t)r * 3 * C + (2 * nH + head) * 32 + d];
    }
    __syncthreads();

    if (tid >= 100) return;

    const float scale = 0.17677669529663687f;  // 32^-0.5
    float q[32];
    const float* qp = base + (size_t)tid * 3 * C + head * 32;
#pragma unroll
    for (int d = 0; d < 32; ++d) q[d] = qp[d] * scale;

    const int ri = tid / 10, ci = tid % 10;
    const int Wg = W / 10;
    const int wy = win / Wg, wx = win % Wg;
    int lab_i = 0;
    if (shift) lab_i = region_id(wy * 10 + ri, H) * 3 + region_id(wx * 10 + ci, W);

    float sc[100];
    float mx = -3.4e38f;
    for (int j = 0; j < 100; ++j) {
        float s = 0.0f;
#pragma unroll
        for (int d = 0; d < 32; ++d) s += q[d] * ks[j][d];
        const int rj = j / 10, cj = j % 10;
        s += rpb[((ri - rj + 9) * 19 + (ci - cj + 9)) * nH + head];
        if (shift) {
            const int lab_j = region_id(wy * 10 + rj, H) * 3 + region_id(wx * 10 + cj, W);
            if (lab_j != lab_i) s -= 100.0f;
        }
        sc[j] = s;
        mx = fmaxf(mx, s);
    }
    float sum = 0.0f;
    for (int j = 0; j < 100; ++j) { sc[j] = expf(sc[j] - mx); sum += sc[j]; }
    const float inv = 1.0f / sum;

    float o[32];
#pragma unroll
    for (int d = 0; d < 32; ++d) o[d] = 0.0f;
    for (int j = 0; j < 100; ++j) {
        const float p = sc[j] * inv;
#pragma unroll
        for (int d = 0; d < 32; ++d) o[d] += p * vs[j][d];
    }
    __bf16* op = out + ((size_t)win * 100 + tid) * C + head * 32;
#pragma unroll
    for (int d = 0; d < 32; ++d) op[d] = (__bf16)o[d];
}

// ---------------------------------------------------------------------------
// Patch embed: 4x4 stride-4 conv, x(3,640,640) -> tokens (160*160, 128)
// ---------------------------------------------------------------------------
__global__ __launch_bounds__(256)
void patch_embed_kernel(const float* __restrict__ x, const float* __restrict__ w,
                        const float* __restrict__ b, float* __restrict__ out)
{
    const int i = blockIdx.x * 256 + threadIdx.x;
    if (i >= 160 * 160 * 128) return;
    const int oc = i % 128;
    const int t = i / 128;
    const int h = t / 160, ww = t % 160;
    float s = b[oc];
#pragma unroll
    for (int ci = 0; ci < 3; ++ci)
#pragma unroll
        for (int kh = 0; kh < 4; ++kh)
#pragma unroll
            for (int kw = 0; kw < 4; ++kw)
                s += x[(size_t)ci * 640 * 640 + (size_t)(h * 4 + kh) * 640 + (ww * 4 + kw)]
                   * w[(((size_t)oc * 3 + ci) * 4 + kh) * 4 + kw];
    out[(size_t)t * 128 + oc] = s;
}

// ---------------------------------------------------------------------------
// Patch merging gather: x[H*W,C] -> xc[(H/2)*(W/2), 4C] (fp32; LN follows)
// quadrant order: (0,0),(1,0),(0,1),(1,1)
// ---------------------------------------------------------------------------
__global__ __launch_bounds__(256)
void merge_gather_kernel(const float* __restrict__ x, float* __restrict__ xc,
                         int H, int W, int C)
{
    const int C4 = 4 * C;
    const int total = (H / 2) * (W / 2) * C4;
    const int i = blockIdx.x * 256 + threadIdx.x;
    if (i >= total) return;
    const int j = i % C4;
    const int row = i / C4;
    const int q = j / C, c = j % C;
    const int W2 = W / 2;
    const int h2 = row / W2, w2 = row % W2;
    const int dh = q & 1, dw = q >> 1;
    xc[i] = x[((size_t)(2 * h2 + dh) * W + (2 * w2 + dw)) * C + c];
}

__global__ __launch_bounds__(256)
void add_kernel(float* __restrict__ x, const float* __restrict__ y, int n)
{
    const int i = blockIdx.x * 256 + threadIdx.x;
    if (i < n) x[i] += y[i];
}

// ---------------------------------------------------------------------------
// Host orchestration
// ---------------------------------------------------------------------------
static inline void launch_gemm(const __bf16* A, const float* Bw, const float* bias,
                               void* C, int M, int N, int K, int epi, int out_bf16,
                               __bf16* bfB, hipStream_t s)
{
    // weights: fp32 [K,N] -> bf16 transposed [N,K]
    dim3 tg((N + 31) / 32, (K + 31) / 32);
    convT_kernel<<<tg, 256, 0, s>>>(Bw, bfB, K, N);
    dim3 grid((N + GT_N - 1) / GT_N, (M + GT_M - 1) / GT_M);
    gemm_wmma_bf16<<<grid, 256, 0, s>>>(A, bfB, bias, C, M, N, K, epi, out_bf16);
}

extern "C" void kernel_launch(void* const* d_in, const int* in_sizes, int n_in,
                              void* d_out, int out_size, void* d_ws, size_t ws_size,
                              hipStream_t stream)
{
    (void)in_sizes; (void)n_in; (void)out_size; (void)ws_size;
    static const int DEPTHS[4] = {2, 2, 14, 2};
    static const int HEADS_[4] = {4, 8, 16, 32};

    int idx = 0;
    auto F = [&](void) -> const float* { return (const float*)d_in[idx++]; };

    const float* X     = F();
    const float* pe_w  = F();
    const float* pe_b  = F();
    const float* pe_ng = F();
    const float* pe_nb = F();

    // Workspace carve. Stage-0 sizes dominate.
    const size_t S   = (size_t)25600 * 128;   // 3.27M elems (tokens * C at stage 0)
    const size_t SZ2 = (size_t)25600 * 512;   // qkv / mlp-hidden max elems
    float*  buf0 = (float*)d_ws;              // persistent tokens x (fp32)
    float*  buf1 = buf0 + S;                  // fp32 temporaries (LN out, proj out)
    float*  buf3 = buf1 + S;                  // fp32/bf16 temporaries (win, attn, mlp out)
    float*  buf2 = buf3 + S;                  // qkv fp32 / mlp hidden bf16
    __bf16* bfB  = (__bf16*)(buf2 + SZ2);     // transposed bf16 weights (max 4.2M elems)

    // patch embed -> buf1, LN -> buf0 (fp32 residual stream)
    {
        const int total = 160 * 160 * 128;
        patch_embed_kernel<<<(total + 255) / 256, 256, 0, stream>>>(X, pe_w, pe_b, buf1);
        ln_kernel<<<160 * 160, 256, 0, stream>>>(buf1, pe_ng, pe_nb, buf0, nullptr, 128);
    }

    int H = 160, W = 160, C = 128;
    for (int s = 0; s < 4; ++s) {
        const int nH = HEADS_[s];
        const int L = H * W;
        const int nW = (H / 10) * (W / 10);
        const int elems = L * C;
        const int eb = (elems + 255) / 256;

        for (int b = 0; b < DEPTHS[s]; ++b) {
            const float* n1g = F();  const float* n1b = F();
            const float* qkv_w = F(); const float* qkv_b = F();
            const float* rpb = F();
            const float* proj_w = F(); const float* proj_b = F();
            const float* n2g = F();  const float* n2b = F();
            const float* fc1_w = F(); const float* fc1_b = F();
            const float* fc2_w = F(); const float* fc2_b = F();
            const int shift = (b % 2) ? 5 : 0;

            // LN1 (fp32 out; feeds the shifted-window gather)
            ln_kernel<<<L, 256, 0, stream>>>(buf0, n1g, n1b, buf1, nullptr, C);
            // window partition (+shift) -> bf16 tokens
            win_part_kernel<<<eb, 256, 0, stream>>>(buf1, (__bf16*)buf3, H, W, C, shift);
            // qkv GEMM (fp32 out; attention reads fp32)
            launch_gemm((const __bf16*)buf3, qkv_w, qkv_b, buf2, L, 3 * C, C,
                        EPI_BIAS, 0, bfB, stream);
            // attention -> bf16
            attn_kernel<<<dim3(nW, nH), 128, 0, stream>>>(buf2, rpb, (__bf16*)buf3,
                                                          nH, H, W, shift);
            // proj GEMM (fp32 out; feeds window-reverse + residual)
            launch_gemm((const __bf16*)buf3, proj_w, proj_b, buf1, L, C, C,
                        EPI_BIAS, 0, bfB, stream);
            win_rev_add_kernel<<<eb, 256, 0, stream>>>(buf1, buf0, H, W, C, shift);
            // LN2 -> bf16 (feeds fc1)
            ln_kernel<<<L, 256, 0, stream>>>(buf0, n2g, n2b, nullptr, (__bf16*)buf1, C);
            // MLP: fc1 (bf16 out, gelu) -> fc2 (fp32 out)
            launch_gemm((const __bf16*)buf1, fc1_w, fc1_b, buf2, L, 4 * C, C,
                        EPI_GELU, 1, bfB, stream);
            launch_gemm((const __bf16*)buf2, fc2_w, fc2_b, buf3, L, C, 4 * C,
                        EPI_BIAS, 0, bfB, stream);
            add_kernel<<<eb, 256, 0, stream>>>(buf0, buf3, elems);
        }

        if (s < 3) {
            const float* ds_ng = F(); const float* ds_nb = F(); const float* ds_w = F();
            merge_gather_kernel<<<eb, 256, 0, stream>>>(buf0, buf1, H, W, C);
            H /= 2; W /= 2;
            // LN over 4C -> bf16 (feeds merge GEMM)
            ln_kernel<<<H * W, 256, 0, stream>>>(buf1, ds_ng, ds_nb, nullptr,
                                                 (__bf16*)buf3, 4 * C);
            launch_gemm((const __bf16*)buf3, ds_w, nullptr, buf0, H * W, 2 * C, 4 * C,
                        EPI_NONE, 0, bfB, stream);
            C *= 2;
        }
    }

    // final tokens: (400, 1024) fp32
    hipMemcpyAsync(d_out, buf0, (size_t)400 * 1024 * sizeof(float),
                   hipMemcpyDeviceToDevice, stream);
}
